// GATModel_40939628265881
// MI455X (gfx1250) — matrix operand order repaired and checked by reading, not compiled, not run on hardware
//
#include <hip/hip_runtime.h>
#include <hip/hip_bf16.h>
#include <math.h>

// ---------------------------------------------------------------------------
// GRU(6->64) -> GRU(64->64) -> BN -> [gram trick] -> FC -> BN -> LReLU -> FC(1)
//
// Roofline (MI455X): collapsed pipeline ~37 GFLOP / ~150 MB traffic.
// HBM floor ~7us @23.3TB/s; dominant cost = T=60 *sequential* recurrent GEMMs
// (h @ W_hh^T : Nx64 @ 64x192 per step) -> v_wmma_f32_16x16x32_bf16, with
// time-invariant W_hh fragments pinned in VGPRs, interleaved accumulator
// chains (avoid WMMA->WMMA RAW NOPs), and h carried in registers per lane.
// The naive gamma(8192x8192) is algebraically collapsed to a 64x64 Gram.
// Workspace needed: ~68.3 MB.
// ---------------------------------------------------------------------------

#define DFEAT 6
#define TT    60
#define HH    64
#define NN    8192
#define G3    192          // 3*H gate width
#define LRELU 0.01f
#define EPS   1e-5f

typedef __bf16 bf16_t;
typedef __attribute__((ext_vector_type(16))) __bf16 bf16x16;
typedef __attribute__((ext_vector_type(8)))  __bf16 bf16x8;
typedef __attribute__((ext_vector_type(8)))  float  f32x8;

__device__ __forceinline__ f32x8 wmma_bf16(bf16x16 a, bf16x16 b, f32x8 c) {
  // D = A(16x32) * B(32x16) + C, f32 accumulate
  return __builtin_amdgcn_wmma_f32_16x16x32_bf16(false, a, false, b, (short)0, c,
                                                 false, false);
}

// A-fragment, 16x32 bf16, row-major source with row stride ld (elements).
// CDNA5 layout: lane<16 -> row=lane, K {0..7, 16..23}; lane>=16 -> row=lane-16,
// K {8..15, 24..31}. Two 16B loads per lane.
__device__ __forceinline__ bf16x16 load_fragA(const bf16_t* base, int ld) {
  const int lane = threadIdx.x & 31;
  const int m  = lane & 15;
  const int kb = (lane >> 4) << 3;              // 0 or 8
  const bf16_t* p = base + (size_t)m * ld + kb;
  bf16x8 lo = *reinterpret_cast<const bf16x8*>(p);
  bf16x8 hi = *reinterpret_cast<const bf16x8*>(p + 16);
  bf16x16 f;
#pragma unroll
  for (int i = 0; i < 8; ++i) { f[i] = lo[i]; f[i + 8] = hi[i]; }
  return f;
}

// B-fragment, 32x16 bf16 (KxN). Source stored row-major as [n][k] (i.e. W with
// output-gate rows). Lane<16 -> col n=lane, K {0..7, 16..23}; lane>=16 -> K {8..15, 24..31}.
__device__ __forceinline__ bf16x16 load_fragB(const bf16_t* base, int ld) {
  const int lane = threadIdx.x & 31;
  const int n  = lane & 15;
  const int kb = (lane >> 4) << 3;              // 0 or 8
  const bf16_t* p = base + (size_t)n * ld + kb;
  bf16x8 lo = *reinterpret_cast<const bf16x8*>(p);
  bf16x8 hi = *reinterpret_cast<const bf16x8*>(p + 16);
  bf16x16 f;
#pragma unroll
  for (int i = 0; i < 8; ++i) { f[i] = lo[i]; f[i + 8] = hi[i]; }
  return f;
}

// ---------------------------------------------------------------------------
// Fused GRU layer. 32 batch rows per 128-thread block (4 wave32 waves).
// Wave w: row-tile rt = w&1 (16 rows), col-groups {2*(w>>1), 2*(w>>1)+1}.
// W_hh fragments live in VGPRs across all 60 steps; h state per lane lives in
// registers (LDS copy kept only for the cross-wave WMMA A-fragment exchange).
// ---------------------------------------------------------------------------
template <int LAYER>
__global__ __launch_bounds__(128) void gru_kernel(
    const float* __restrict__ x,         // LAYER0: (N, 6*60)
    const bf16_t* __restrict__ in_seq,   // LAYER1: (N, T, 64) bf16
    const float* __restrict__ W_ih,      // (192, in)
    const float* __restrict__ W_hh,      // (192, 64)
    const float* __restrict__ b_ih, const float* __restrict__ b_hh,
    bf16_t* __restrict__ seq_out,        // LAYER0: (N, T, 64) bf16
    float* __restrict__ hidden_out)      // LAYER1: (N, 64) f32 (last step)
{
  constexpr int ROWS = 32;
  __shared__ __align__(16) bf16_t Whh_s[G3 * HH];
  __shared__ __align__(16) bf16_t Wih_s[(LAYER == 1) ? G3 * HH : 8];
  __shared__ float  Wih0_s[(LAYER == 0) ? G3 * DFEAT : 1];
  __shared__ float  xfull_s[(LAYER == 0) ? ROWS * DFEAT * TT : 1];
  __shared__ __align__(16) bf16_t h_s[ROWS * HH];
  __shared__ float  bih_s[G3], bhh_s[G3];

  const int tid = threadIdx.x;
  const int rowbase = blockIdx.x * ROWS;

  for (int i = tid; i < G3 * HH; i += 128) Whh_s[i] = (bf16_t)W_hh[i];
  if (LAYER == 1)
    for (int i = tid; i < G3 * HH; i += 128) Wih_s[i] = (bf16_t)W_ih[i];
  if (LAYER == 0) {
    for (int i = tid; i < G3 * DFEAT; i += 128) Wih0_s[i] = W_ih[i];
    // stage the full x tile once (coalesced), instead of 60 strided passes
    for (int i = tid; i < ROWS * DFEAT * TT; i += 128) {
      const int r = i / (DFEAT * TT), q = i % (DFEAT * TT);
      xfull_s[i] = x[(size_t)(rowbase + r) * (DFEAT * TT) + q];
    }
  }
  for (int i = tid; i < G3; i += 128) { bih_s[i] = b_ih[i]; bhh_s[i] = b_hh[i]; }
  for (int i = tid; i < ROWS * HH; i += 128) h_s[i] = (bf16_t)0.0f;
  __syncthreads();

  const int wave = tid >> 5;
  const int lane = tid & 31;
  const int rt   = wave & 1;              // 16-row tile within the 32-row block
  const int cg0  = (wave >> 1) * 2;       // first of two 16-col hidden groups
  const int nloc = lane & 15;             // C/D: col within tile
  const int mloc = rt * 16 + ((lane >> 4) << 3);  // C/D: first row (of 8)

  // ---- time-invariant per-wave state pinned in registers ----
  bf16x16 bh[2][3][2];                    // [pair][gate][khalf] W_hh B-frags
#pragma unroll
  for (int p = 0; p < 2; ++p)
#pragma unroll
    for (int g = 0; g < 3; ++g)
#pragma unroll
      for (int kh = 0; kh < 2; ++kh)
        bh[p][g][kh] =
            load_fragB(Whh_s + (size_t)(g * HH + (cg0 + p) * 16) * HH + kh * 32, HH);

  float brc[2], bzc[2], bnic[2], bnhc[2];
  float wr[2][DFEAT], wz[2][DFEAT], wn[2][DFEAT];
#pragma unroll
  for (int p = 0; p < 2; ++p) {
    const int j = (cg0 + p) * 16 + nloc;
    brc[p]  = bih_s[j] + bhh_s[j];
    bzc[p]  = bih_s[HH + j] + bhh_s[HH + j];
    bnic[p] = bih_s[2 * HH + j];
    bnhc[p] = bhh_s[2 * HH + j];
    if (LAYER == 0) {
#pragma unroll
      for (int i = 0; i < DFEAT; ++i) {
        wr[p][i] = Wih0_s[j * DFEAT + i];
        wz[p][i] = Wih0_s[(HH + j) * DFEAT + i];
        wn[p][i] = Wih0_s[(2 * HH + j) * DFEAT + i];
      }
    }
  }

  float hcur[2][8];                       // h state owned by this lane
#pragma unroll
  for (int p = 0; p < 2; ++p)
#pragma unroll
    for (int k = 0; k < 8; ++k) hcur[p][k] = 0.f;

  for (int t = 0; t < TT; ++t) {
    // ---- phase A: read h_s (A-fragments) + stream x A-fragments ----
    bf16x16 ah0 = load_fragA(h_s + rt * 16 * HH, HH);        // K 0..31
    bf16x16 ah1 = load_fragA(h_s + rt * 16 * HH + 32, HH);   // K 32..63
    bf16x16 ax0 = {}, ax1 = {};
    if (LAYER == 1) {
      const bf16_t* xb = in_seq + (size_t)(rowbase + rt * 16) * (TT * HH) + t * HH;
      ax0 = load_fragA(xb, TT * HH);
      ax1 = load_fragA(xb + 32, TT * HH);
    }
    __syncthreads();

    // ---- phase B: WMMA gates (interleaved chains) + elementwise update ----
#pragma unroll
    for (int p = 0; p < 2; ++p) {
      const int jb = (cg0 + p) * 16;
      f32x8 accR = {}, accZ = {}, accNh = {}, accNx = {};
      accR  = wmma_bf16(ah0, bh[p][0][0], accR);
      accZ  = wmma_bf16(ah0, bh[p][1][0], accZ);
      accNh = wmma_bf16(ah0, bh[p][2][0], accNh);
      accR  = wmma_bf16(ah1, bh[p][0][1], accR);
      accZ  = wmma_bf16(ah1, bh[p][1][1], accZ);
      accNh = wmma_bf16(ah1, bh[p][2][1], accNh);
      if (LAYER == 1) {
        // batch the 6 W_ih fragment loads -> one clause, one wait
        bf16x16 bi00 = load_fragB(Wih_s + (size_t)jb * HH, HH);
        bf16x16 bi01 = load_fragB(Wih_s + (size_t)jb * HH + 32, HH);
        bf16x16 bi10 = load_fragB(Wih_s + (size_t)(HH + jb) * HH, HH);
        bf16x16 bi11 = load_fragB(Wih_s + (size_t)(HH + jb) * HH + 32, HH);
        bf16x16 bi20 = load_fragB(Wih_s + (size_t)(2 * HH + jb) * HH, HH);
        bf16x16 bi21 = load_fragB(Wih_s + (size_t)(2 * HH + jb) * HH + 32, HH);
        accR  = wmma_bf16(ax0, bi00, accR);
        accZ  = wmma_bf16(ax0, bi10, accZ);
        accNx = wmma_bf16(ax0, bi20, accNx);
        accR  = wmma_bf16(ax1, bi01, accR);
        accZ  = wmma_bf16(ax1, bi11, accZ);
        accNx = wmma_bf16(ax1, bi21, accNx);
      }
      const int j = jb + nloc;
#pragma unroll
      for (int k = 0; k < 8; ++k) {
        const int rloc = mloc + k;
        float xr = 0.f, xz = 0.f, xn = 0.f;
        if (LAYER == 0) {
#pragma unroll
          for (int i = 0; i < DFEAT; ++i) {
            const float xv = xfull_s[rloc * (DFEAT * TT) + i * TT + t];
            xr += xv * wr[p][i];
            xz += xv * wz[p][i];
            xn += xv * wn[p][i];
          }
        } else {
          xn = accNx[k];
        }
        const float rg = 1.f / (1.f + __expf(-(accR[k] + xr + brc[p])));
        const float zg = 1.f / (1.f + __expf(-(accZ[k] + xz + bzc[p])));
        const float ng = tanhf(xn + bnic[p] + rg * (accNh[k] + bnhc[p]));
        const float hv = (1.f - zg) * ng + zg * hcur[p][k];
        hcur[p][k] = hv;
        h_s[rloc * HH + j] = (bf16_t)hv;
        if (LAYER == 0) {
          seq_out[((size_t)(rowbase + rloc) * TT + t) * HH + j] = (bf16_t)hv;
        } else if (t == TT - 1) {
          hidden_out[(size_t)(rowbase + rloc) * HH + j] = hv;
        }
      }
    }
    if (LAYER == 1 && t + 1 < TT) {
      // prefetch next step's A rows (global_prefetch_b8)
      __builtin_prefetch(
          (const void*)(in_seq +
                        (size_t)(rowbase + rt * 16 + (lane & 15)) * (TT * HH) +
                        (t + 1) * HH),
          0, 0);
    }
    __syncthreads();
  }
}

// ---------------------------------------------------------------------------
// Small post-GRU pipeline
// ---------------------------------------------------------------------------
__global__ void zero_kernel(float* p, int n) {
  int i = blockIdx.x * blockDim.x + threadIdx.x;
  if (i < n) p[i] = 0.f;
}

// Per-column sum / sumsq over N rows of an (N,64) f32 matrix (atomics).
__global__ __launch_bounds__(256) void colstats_kernel(
    const float* __restrict__ X, int n, float* sum, float* sumsq) {
  const int j = threadIdx.x & 63;
  const int rpb = 256 >> 6;  // 4 rows per block-pass
  float s = 0.f, q = 0.f;
  for (int i = blockIdx.x * rpb + (threadIdx.x >> 6); i < n; i += gridDim.x * rpb) {
    const float v = X[(size_t)i * 64 + j];
    s += v; q += v * v;
  }
  atomicAdd(&sum[j], s);
  atomicAdd(&sumsq[j], q);
}

__global__ void bnfin_kernel(const float* sum, const float* sumsq, int n,
                             float* mean, float* rstd) {
  const int j = threadIdx.x;
  if (j < 64) {
    const float m = sum[j] / n;
    const float v = sumsq[j] / n - m * m;
    mean[j] = m;
    rstd[j] = rsqrtf(v + EPS);
  }
}

// BN1 normalize -> hb (bf16); row means e_n; partial u += e_n * hb[n,:]
__global__ __launch_bounds__(256) void hb_kernel(
    const float* __restrict__ hidden, const float* mean, const float* rstd,
    const float* bw, const float* bb, bf16_t* __restrict__ hb,
    float* __restrict__ e_, float* u) {
  __shared__ float red[4][64];
  __shared__ float erow[4];
  const int r = threadIdx.x >> 6, j = threadIdx.x & 63;
  const size_t n = (size_t)blockIdx.x * 4 + r;
  const float v = (hidden[n * 64 + j] - mean[j]) * rstd[j] * bw[j] + bb[j];
  hb[n * 64 + j] = (bf16_t)v;
  red[r][j] = v;
  __syncthreads();
  if (j == 0) {
    float s = 0.f;
    for (int c = 0; c < 64; ++c) s += red[r][c];
    const float en = s * (1.f / 64.f);
    e_[n] = en;
    erow[r] = en;
  }
  __syncthreads();
  if (r == 0) {
    float acc = 0.f;
    for (int rr = 0; rr < 4; ++rr) acc += erow[rr] * red[rr][j];
    atomicAdd(&u[j], acc);
  }
}

// G = hb^T @ hb (64x64), tiled over 64-row blocks, atomic accumulate.
__global__ __launch_bounds__(256) void gram_kernel(const bf16_t* __restrict__ hb,
                                                   float* G) {
  __shared__ float tile[64][65];
  const int tid = threadIdx.x;
  const int i  = tid >> 2;
  const int jb = (tid & 3) * 16;
  float acc[16];
#pragma unroll
  for (int k = 0; k < 16; ++k) acc[k] = 0.f;
  for (int rb = blockIdx.x; rb < NN / 64; rb += gridDim.x) {
    __syncthreads();
    for (int idx = tid; idx < 64 * 64; idx += 256) {
      tile[idx >> 6][idx & 63] = (float)hb[(size_t)(rb * 64 + (idx >> 6)) * 64 + (idx & 63)];
    }
    __syncthreads();
    for (int rr = 0; rr < 64; ++rr) {
      const float a = tile[rr][i];
#pragma unroll
      for (int k = 0; k < 16; ++k) acc[k] += a * tile[rr][jb + k];
    }
  }
#pragma unroll
  for (int k = 0; k < 16; ++k) atomicAdd(&G[i * 64 + jb + k], acc[k]);
}

// Mt[j][k'] = (1/H) * sum_k G[k'][k] * fc_w[j][k]   (stored transposed, bf16)
// v[j]      = sum_k u[k] * fc_w[j][k]
__global__ __launch_bounds__(256) void mt_kernel(const float* G, const float* u,
                                                 const float* __restrict__ fcw,
                                                 bf16_t* Mt, float* v) {
  const int j   = threadIdx.x >> 2;
  const int kp0 = (threadIdx.x & 3) * 16;
  for (int kk = 0; kk < 16; ++kk) {
    const int kp = kp0 + kk;
    float s = 0.f;
    for (int k = 0; k < 64; ++k) s += G[kp * 64 + k] * fcw[j * 64 + k];
    Mt[j * 64 + kp] = (bf16_t)(s * (1.f / 64.f));
  }
  if ((threadIdx.x & 3) == 0) {
    float s = 0.f;
    for (int k = 0; k < 64; ++k) s += u[k] * fcw[j * 64 + k];
    v[j] = s;
  }
}

// out2[n,:] = hb[n,:] @ Mt - e_n * v + fc_b   (WMMA GEMM, 64 rows / block)
__global__ __launch_bounds__(128) void out2_kernel(
    const bf16_t* __restrict__ hb, const bf16_t* __restrict__ Mt,
    const float* __restrict__ e_, const float* v, const float* fcb,
    float* __restrict__ out2) {
  __shared__ __align__(16) bf16_t Mt_s[64 * 64];
  const int tid = threadIdx.x;
  for (int i = tid; i < 64 * 64; i += 128) Mt_s[i] = Mt[i];
  __syncthreads();
  const int wave = tid >> 5, lane = tid & 31;
  const int rowbase = blockIdx.x * 64 + wave * 16;
  const int nloc = lane & 15, moff = (lane >> 4) << 3;
  bf16x16 a0 = load_fragA(hb + (size_t)rowbase * 64, 64);
  bf16x16 a1 = load_fragA(hb + (size_t)rowbase * 64 + 32, 64);
  bf16x16 b0[4], b1[4];
#pragma unroll
  for (int ct = 0; ct < 4; ++ct) {
    b0[ct] = load_fragB(Mt_s + (size_t)(ct * 16) * 64, 64);
    b1[ct] = load_fragB(Mt_s + (size_t)(ct * 16) * 64 + 32, 64);
  }
  f32x8 acc[4] = {{}, {}, {}, {}};
#pragma unroll
  for (int ct = 0; ct < 4; ++ct) acc[ct] = wmma_bf16(a0, b0[ct], acc[ct]);
#pragma unroll
  for (int ct = 0; ct < 4; ++ct) acc[ct] = wmma_bf16(a1, b1[ct], acc[ct]);
#pragma unroll
  for (int ct = 0; ct < 4; ++ct) {
    const int j = ct * 16 + nloc;
    const float vj = v[j], bj = fcb[j];
#pragma unroll
    for (int k = 0; k < 8; ++k) {
      const size_t n = rowbase + moff + k;
      out2[n * 64 + j] = acc[ct][k] - e_[n] * vj + bj;
    }
  }
}

// BN2 + LeakyReLU + dot with fc_out_w -> scalar per row.
__global__ __launch_bounds__(256) void final_kernel(
    const float* __restrict__ out2, const float* mean2, const float* rstd2,
    const float* bw, const float* bb, const float* fow, const float* fob,
    float* __restrict__ out) {
  __shared__ float red[4][64];
  const int r = threadIdx.x >> 6, j = threadIdx.x & 63;
  const size_t n = (size_t)blockIdx.x * 4 + r;
  float v = (out2[n * 64 + j] - mean2[j]) * rstd2[j] * bw[j] + bb[j];
  v = (v >= 0.f) ? v : LRELU * v;
  red[r][j] = v * fow[j];
  __syncthreads();
  if (j == 0) {
    float s = 0.f;
    for (int c = 0; c < 64; ++c) s += red[r][c];
    out[n] = s + fob[0];
  }
}

// ---------------------------------------------------------------------------
extern "C" void kernel_launch(void* const* d_in, const int* in_sizes, int n_in,
                              void* d_out, int out_size, void* d_ws, size_t ws_size,
                              hipStream_t stream) {
  const float* x    = (const float*)d_in[0];
  const float* Wih0 = (const float*)d_in[1];
  const float* Whh0 = (const float*)d_in[2];
  const float* bih0 = (const float*)d_in[3];
  const float* bhh0 = (const float*)d_in[4];
  const float* Wih1 = (const float*)d_in[5];
  const float* Whh1 = (const float*)d_in[6];
  const float* bih1 = (const float*)d_in[7];
  const float* bhh1 = (const float*)d_in[8];
  const float* bn1w = (const float*)d_in[9];
  const float* bn1b = (const float*)d_in[10];
  const float* fcw  = (const float*)d_in[11];
  const float* fcb  = (const float*)d_in[12];
  const float* bn2w = (const float*)d_in[13];
  const float* bn2b = (const float*)d_in[14];
  const float* fow  = (const float*)d_in[15];
  const float* fob  = (const float*)d_in[16];
  float* out = (float*)d_out;

  char* ws = (char*)d_ws;
  size_t off = 0;
  bf16_t* hseq   = (bf16_t*)(ws + off); off += (size_t)NN * TT * HH * 2;  // 63 MB
  float*  hidden = (float*)(ws + off);  off += (size_t)NN * HH * 4;
  bf16_t* hb     = (bf16_t*)(ws + off); off += (size_t)NN * HH * 2;
  float*  e_     = (float*)(ws + off);  off += (size_t)NN * 4;
  float*  out2   = (float*)(ws + off);  off += (size_t)NN * HH * 4;
  float*  st     = (float*)(ws + off);  off += 4736 * 4;
  bf16_t* Mt     = (bf16_t*)(ws + off); off += 4096 * 2;
  float *bn1sum = st,       *bn1sq = st + 64,  *u = st + 128,
        *bn2sum = st + 192, *bn2sq = st + 256,
        *mean1 = st + 320,  *rstd1 = st + 384, *vv = st + 448,
        *mean2 = st + 512,  *rstd2 = st + 576, *G  = st + 640;

  zero_kernel<<<(4736 + 255) / 256, 256, 0, stream>>>(st, 4736);
  gru_kernel<0><<<NN / 32, 128, 0, stream>>>(x, nullptr, Wih0, Whh0, bih0, bhh0,
                                             hseq, nullptr);
  gru_kernel<1><<<NN / 32, 128, 0, stream>>>(nullptr, hseq, Wih1, Whh1, bih1, bhh1,
                                             nullptr, hidden);
  colstats_kernel<<<128, 256, 0, stream>>>(hidden, NN, bn1sum, bn1sq);
  bnfin_kernel<<<1, 64, 0, stream>>>(bn1sum, bn1sq, NN, mean1, rstd1);
  hb_kernel<<<NN / 4, 256, 0, stream>>>(hidden, mean1, rstd1, bn1w, bn1b, hb, e_, u);
  gram_kernel<<<64, 256, 0, stream>>>(hb, G);
  mt_kernel<<<1, 256, 0, stream>>>(G, u, fcw, Mt, vv);
  out2_kernel<<<NN / 64, 128, 0, stream>>>(hb, Mt, e_, vv, fcb, out2);
  colstats_kernel<<<128, 256, 0, stream>>>(out2, NN, bn2sum, bn2sq);
  bnfin_kernel<<<1, 64, 0, stream>>>(bn2sum, bn2sq, NN, mean2, rstd2);
  final_kernel<<<NN / 4, 256, 0, stream>>>(out2, mean2, rstd2, bn2w, bn2b, fow, fob, out);
}
